// CNN_SIMPLE_78769700209049
// MI455X (gfx1250) — matrix operand order repaired and checked by reading, not compiled, not run on hardware
//
#include <hip/hip_runtime.h>
#include <hip/hip_bf16.h>

// ---------------------------------------------------------------------------
// MI455X (gfx1250) implementation, wave32 WMMA.
// Convs lowered to implicit GEMM on v_wmma_f32_16x16x32_bf16, activations
// NHWC bf16, f32 accumulation, fused bias+ReLU. LDS staged with b128 vector
// ops: A tiles K-major, B tiles N-major so every fragment is 2x ds_load_b128.
// ---------------------------------------------------------------------------

typedef __bf16 bf16_t;
typedef __attribute__((ext_vector_type(16))) __bf16 v16bf;
typedef __attribute__((ext_vector_type(8)))  __bf16 v8bf;   // 16 bytes
typedef __attribute__((ext_vector_type(8)))  float  v8f;

#define WMMA_BF16(a, b, c) \
  __builtin_amdgcn_wmma_f32_16x16x32_bf16(false, (a), false, (b), (short)0, (c), false, false)

#define CAT16(lo, hi) \
  __builtin_shufflevector((lo), (hi), 0, 1, 2, 3, 4, 5, 6, 7, 8, 9, 10, 11, 12, 13, 14, 15)

// ---- WMMA fragment loaders from LDS (layouts per CDNA5 ISA 7.12.2) --------

// A: 16x32 bf16 tile, K-major rows with stride `lds` (elements, *2B must be
// 16B-multiple). lane<16: row M=lane, K={0..7,16..23}; lane>=16: K={8..15,24..31}.
__device__ inline v16bf load_frag_a(const bf16_t* base, int lds, int lane) {
  int row   = lane & 15;
  int khalf = (lane >> 4) << 3;  // 0 or 8
  const v8bf* p = (const v8bf*)(base + row * lds + khalf);
  return CAT16(p[0], p[2]);      // K = khalf+e  and  khalf+16+e
}

// B: 32x16 bf16 tile from an N-major LDS buffer Bs[n][k] with K-stride `ldk`.
// lane<16: col N=lane, K={0..7,16..23} ; lane>=16: col N=lane-16, K={8..15,24..31}.
__device__ inline v16bf load_frag_b(const bf16_t* Bs, int ldk, int n0,
                                    int lane) {
  int col   = lane & 15;
  int khalf = (lane >> 4) << 3;  // 0 or 8
  const v8bf* p = (const v8bf*)(Bs + (n0 + col) * ldk + khalf);
  return CAT16(p[0], p[2]);
}

// ---------------------------------------------------------------------------
// Implicit-GEMM conv: y[b,ho,wo,co] = relu(sum_{kh,kw,ci} x[b,2ho+kh,2wo+kw,ci]
//                                          * w[co,ci,kh,kw] + bias[co])
// x NHWC bf16, wT reordered [(kh*3+kw)*CIN+ci, COUT] bf16, y NHWC bf16.
// SAME pad, stride 2, k=3, even H/W => pad_beg=0 (ih=2*ho+kh, guard < HIN).
// Block tile 256(M) x 64(N), K-step 32; 8 waves, each 32x64 via 2x4 WMMA.
// ---------------------------------------------------------------------------
template <int CIN, int COUT, int HIN, int WIN>
__global__ __launch_bounds__(256) void conv_wmma_kernel(
    const bf16_t* __restrict__ x, const bf16_t* __restrict__ wT,
    const float* __restrict__ bias, bf16_t* __restrict__ y) {
  constexpr int HOUT = HIN / 2, WOUT = WIN / 2;
  constexpr int KT   = 9 * CIN;
  constexpr int BM = 256, BN = 64, BK = 32;
  constexpr int LDA  = BK + 8;  // 40 elems = 80B rows (16B multiple)
  constexpr int LDBK = BK + 8;  // N-major B rows
  constexpr int MAXM = 64 * HOUT * WOUT;

  __shared__ __align__(16) bf16_t Asm[BM * LDA];
  __shared__ __align__(16) bf16_t Bsm[BN * LDBK];

  const int tid  = threadIdx.x;
  const int lane = tid & 31;
  const int wave = tid >> 5;
  const int m0   = blockIdx.x * BM;
  const int n0   = blockIdx.y * BN;
  const int wm   = wave * 32;  // wave M offset inside block tile

  v8f acc[2][4] = {};

  // Per-thread A row decode (thread -> one m row, all 32 k's of the step).
  const int mi = tid;  // 0..255
  const int m  = m0 + mi;
  int b  = m / (HOUT * WOUT);
  int r  = m - b * (HOUT * WOUT);
  int ho = r / WOUT;
  int wo = r - ho * WOUT;

  for (int k0 = 0; k0 < KT; k0 += BK) {
    // ---- stage A (im2col): one row of 32 k's per thread -------------------
    if constexpr (CIN % 32 == 0) {
      // 32-wide K-step lies inside a single (kh,kw) group: 64 contiguous B.
      int kk  = k0 / CIN;            // kh*3+kw (power-of-two CIN => shifts)
      int ci  = k0 - kk * CIN;
      int khh = kk / 3, kww = kk - khh * 3;
      int ih  = 2 * ho + khh;
      int iw  = 2 * wo + kww;
      v8bf z = {};
      v8bf v0 = z, v1 = z, v2 = z, v3 = z;
      if (m < MAXM && ih < HIN && iw < WIN) {
        const v8bf* g =
            (const v8bf*)&x[(((size_t)(b * HIN + ih) * WIN + iw) * CIN) + ci];
        v0 = g[0]; v1 = g[1]; v2 = g[2]; v3 = g[3];
      }
      v8bf* dst = (v8bf*)&Asm[mi * LDA];
      dst[0] = v0; dst[1] = v1; dst[2] = v2; dst[3] = v3;
    } else {
      // Generic path (layer 1: CIN=3, KT=27): guarded scalar gather.
#pragma unroll
      for (int e = 0; e < BK; ++e) {
        int k = k0 + e;
        bf16_t v = (bf16_t)0.f;
        if (k < KT && m < MAXM) {
          int ci  = k % CIN;
          int kk  = k / CIN;
          int khh = kk / 3, kww = kk - khh * 3;
          int ih  = 2 * ho + khh;
          int iw  = 2 * wo + kww;
          if (ih < HIN && iw < WIN)
            v = x[(((b * HIN + ih) * WIN + iw) * CIN) + ci];
        }
        Asm[mi * LDA + e] = v;
      }
    }
    // ---- stage B (weights) into N-major LDS: thread -> (n, 8 k's) ---------
    {
      int nl = tid & 63;            // local n
      int kg = (tid >> 6) * 8;      // 0,8,16,24
      v8bf v = {};
#pragma unroll
      for (int e = 0; e < 8; ++e) {
        int k = k0 + kg + e;
        v[e] = (k < KT) ? wT[(size_t)k * COUT + n0 + nl] : (bf16_t)0.f;
      }
      *(v8bf*)&Bsm[nl * LDBK + kg] = v;
      if (k0 + BK < KT)  // prefetch next weight K-step (global_prefetch_b8)
        __builtin_prefetch(&wT[(size_t)(k0 + BK + kg) * COUT + n0 + nl], 0, 1);
    }
    __syncthreads();

    // ---- compute: 2(M) x 4(N) WMMA per wave -------------------------------
    v16bf af0 = load_frag_a(&Asm[(wm + 0)  * LDA], LDA, lane);
    v16bf af1 = load_frag_a(&Asm[(wm + 16) * LDA], LDA, lane);
#pragma unroll
    for (int j = 0; j < 4; ++j) {
      v16bf bf = load_frag_b(Bsm, LDBK, j * 16, lane);
      acc[0][j] = WMMA_BF16(af0, bf, acc[0][j]);
      acc[1][j] = WMMA_BF16(af1, bf, acc[1][j]);
    }
    __syncthreads();
  }

  // ---- epilogue: bias + relu, scatter NHWC (co contiguous across lanes) ---
  const int colLane = lane & 15;
  const int rowOff  = (lane >> 4) * 8;  // C/D layout: VGPR e -> M=e / 8+e
#pragma unroll
  for (int i = 0; i < 2; ++i) {
#pragma unroll
    for (int j = 0; j < 4; ++j) {
      int   n  = n0 + j * 16 + colLane;  // < COUT by grid construction
      float bv = bias[n];
#pragma unroll
      for (int e = 0; e < 8; ++e) {
        int mm = m0 + wm + i * 16 + rowOff + e;
        if (mm < MAXM) {
          float v = acc[i][j][e] + bv;
          v = v > 0.f ? v : 0.f;
          int bb = mm / (HOUT * WOUT);
          int rr = mm - bb * (HOUT * WOUT);
          int oh = rr / WOUT;
          int ow = rr - oh * WOUT;
          y[(((size_t)(bb * HOUT + oh) * WOUT + ow) * COUT) + n] = (bf16_t)v;
        }
      }
    }
  }
}

// ---------------------------------------------------------------------------
// Generic dense GEMM: C = act(A[MxK] * B[KxN] + bias), A/B bf16, acc f32.
// Block tile 64x64, 8 waves: wave -> 16(M) x 32(N) via 2 WMMA per K-step.
// Requires lda, ldb multiples of 8 (all call sites: 512/1000/1024).
// ---------------------------------------------------------------------------
template <bool RELU, bool OUT_BF16>
__global__ __launch_bounds__(256) void gemm_wmma_kernel(
    const bf16_t* __restrict__ A, int lda, const bf16_t* __restrict__ B,
    int ldb, const float* __restrict__ bias, void* __restrict__ Cv, int ldc,
    int M, int N, int K) {
  constexpr int BM = 64, BN = 64, BK = 32;
  constexpr int LDA = BK + 8, LDBK = BK + 8;
  __shared__ __align__(16) bf16_t Asm[BM * LDA];
  __shared__ __align__(16) bf16_t Bsm[BN * LDBK];

  const int tid  = threadIdx.x;
  const int lane = tid & 31;
  const int wave = tid >> 5;
  const int m0   = blockIdx.x * BM;
  const int n0   = blockIdx.y * BN;
  const int wm   = (wave & 3) * 16;
  const int wn   = (wave >> 2) * 32;

  v8f acc[2] = {};

  for (int k0 = 0; k0 < K; k0 += BK) {
    {  // A: 64x32, one v8bf per thread
      int mi = tid >> 2;
      int kq = (tid & 3) * 8;
      int mg = m0 + mi;
      v8bf v = {};
      if (mg < M && k0 + kq + 8 <= K)
        v = *(const v8bf*)&A[(size_t)mg * lda + k0 + kq];
      *(v8bf*)&Asm[mi * LDA + kq] = v;
    }
    {  // B into N-major LDS: thread -> (n, 8 k's)
      int nl = tid & 63;
      int kg = (tid >> 6) * 8;
      v8bf v = {};
      int n = n0 + nl;
#pragma unroll
      for (int e = 0; e < 8; ++e) {
        int k = k0 + kg + e;
        v[e] = (k < K && n < N) ? B[(size_t)k * ldb + n] : (bf16_t)0.f;
      }
      *(v8bf*)&Bsm[nl * LDBK + kg] = v;
    }
    __syncthreads();
    v16bf af  = load_frag_a(&Asm[wm * LDA], LDA, lane);
    v16bf bf0 = load_frag_b(Bsm, LDBK, wn, lane);
    v16bf bf1 = load_frag_b(Bsm, LDBK, wn + 16, lane);
    acc[0] = WMMA_BF16(af, bf0, acc[0]);
    acc[1] = WMMA_BF16(af, bf1, acc[1]);
    __syncthreads();
  }

  const int colLane = lane & 15;
  const int rowOff  = (lane >> 4) * 8;
#pragma unroll
  for (int j = 0; j < 2; ++j) {
    int n = n0 + wn + j * 16 + colLane;
    if (n >= N) continue;
    float bv = bias ? bias[n] : 0.f;
#pragma unroll
    for (int e = 0; e < 8; ++e) {
      int m = m0 + wm + rowOff + e;
      if (m >= M) continue;
      float v = acc[j][e] + bv;
      if (RELU) v = v > 0.f ? v : 0.f;
      if (OUT_BF16)
        ((bf16_t*)Cv)[(size_t)m * ldc + n] = (bf16_t)v;
      else
        ((float*)Cv)[(size_t)m * ldc + n] = v;
    }
  }
}

// ---------------------------------------------------------------------------
// Helper kernels
// ---------------------------------------------------------------------------

__global__ void cvt_f32_bf16(const float* __restrict__ in,
                             bf16_t* __restrict__ out, int n) {
  int i = blockIdx.x * blockDim.x + threadIdx.x;
  if (i < n) out[i] = (bf16_t)in[i];
}

// images NCHW f32 -> NHWC bf16 (B=64, C=3, H=W=224)
__global__ void cvt_img_nhwc(const float* __restrict__ in,
                             bf16_t* __restrict__ out, int n) {
  int i = blockIdx.x * blockDim.x + threadIdx.x;
  if (i >= n) return;
  int c = i % 3;
  int w = (i / 3) % 224;
  int h = (i / (3 * 224)) % 224;
  int b = i / (3 * 224 * 224);
  out[i] = (bf16_t)in[((b * 3 + c) * 224 + h) * 224 + w];
}

// conv weights [Cout,Cin,3,3] f32 -> [(kh*3+kw)*Cin+ci, Cout] bf16
__global__ void reorder_w(const float* __restrict__ w, bf16_t* __restrict__ wT,
                          int Cin, int Cout, int total) {
  int i = blockIdx.x * blockDim.x + threadIdx.x;
  if (i >= total) return;
  int kt = i / Cout;
  int co = i % Cout;
  int ci = kt % Cin;
  int kk = kt / Cin;
  wT[i] = (bf16_t)w[(co * Cin + ci) * 9 + kk];
}

// global average pool: act4 NHWC [64,14,14,512] bf16 -> feats cols [0,512)
__global__ __launch_bounds__(512) void avgpool_kernel(
    const bf16_t* __restrict__ act, bf16_t* __restrict__ feats) {
  int b = blockIdx.x;
  int c = threadIdx.x;
  float s = 0.f;
  const bf16_t* p = act + (size_t)b * 14 * 14 * 512 + c;
#pragma unroll 4
  for (int i = 0; i < 196; ++i) s += (float)p[i * 512];
  feats[b * 1024 + c] = (bf16_t)(s * (1.f / 196.f));
}

// masked mean of embeddings: meanB[b,d] = (1/len) * sum_{s<len} emb[seq[b,s],d]
__global__ __launch_bounds__(256) void embed_mean_kernel(
    const float* __restrict__ emb, const int* __restrict__ seq,
    const int* __restrict__ lens, bf16_t* __restrict__ meanB) {
  int b = blockIdx.x;
  int len = lens[b];
  if (len < 1) len = 1;
  float inv = 1.f / (float)len;
  const int* row = seq + b * 128;
  for (int d = threadIdx.x; d < 512; d += 256) {
    float s = 0.f;
    for (int t = 0; t < len; ++t) s += emb[(size_t)row[t] * 512 + d];
    meanB[b * 512 + d] = (bf16_t)(s * inv);
  }
}

// ---------------------------------------------------------------------------
extern "C" void kernel_launch(void* const* d_in, const int* in_sizes, int n_in,
                              void* d_out, int out_size, void* d_ws,
                              size_t ws_size, hipStream_t stream) {
  const float* images = (const float*)d_in[0];
  const int*   seqs   = (const int*)d_in[1];
  const int*   lens   = (const int*)d_in[2];
  const float* emb    = (const float*)d_in[4];
  const float* cw1 = (const float*)d_in[5];  const float* cb1 = (const float*)d_in[6];
  const float* cw2 = (const float*)d_in[7];  const float* cb2 = (const float*)d_in[8];
  const float* cw3 = (const float*)d_in[9];  const float* cb3 = (const float*)d_in[10];
  const float* cw4 = (const float*)d_in[11]; const float* cb4 = (const float*)d_in[12];
  const float* rnn_w = (const float*)d_in[13]; const float* rnn_b = (const float*)d_in[14];
  const float* fc1_w = (const float*)d_in[15]; const float* fc1_b = (const float*)d_in[16];
  const float* fc2_w = (const float*)d_in[17]; const float* fc2_b = (const float*)d_in[18];
  const float* clf_w = (const float*)d_in[19]; const float* clf_b = (const float*)d_in[20];
  float* out = (float*)d_out;

  size_t off = 0;
  auto carve = [&](size_t elems) -> bf16_t* {
    bf16_t* p = (bf16_t*)((char*)d_ws + off);
    off = (off + elems * sizeof(bf16_t) + 255) & ~(size_t)255;
    return p;
  };
  bf16_t* imgN  = carve((size_t)64 * 224 * 224 * 3);
  bf16_t* act1  = carve((size_t)64 * 112 * 112 * 64);
  bf16_t* act2  = carve((size_t)64 * 56 * 56 * 128);
  bf16_t* act3  = carve((size_t)64 * 28 * 28 * 256);
  bf16_t* act4  = carve((size_t)64 * 14 * 14 * 512);
  bf16_t* w1T   = carve(27 * 64);
  bf16_t* w2T   = carve(576 * 128);
  bf16_t* w3T   = carve(1152 * 256);
  bf16_t* w4T   = carve(2304 * 512);
  bf16_t* rnnB  = carve(512 * 512);
  bf16_t* fc1B  = carve(1024 * 512);
  bf16_t* fc2B  = carve(512 * 512);
  bf16_t* clfB  = carve(512 * 1000);
  bf16_t* meanB = carve(64 * 512);
  bf16_t* feats = carve(64 * 1024);
  bf16_t* h1B   = carve(64 * 512);
  bf16_t* h2B   = carve(64 * 512);
  (void)ws_size; (void)n_in; (void)in_sizes; (void)out_size;

  {
    int n = 64 * 3 * 224 * 224;
    cvt_img_nhwc<<<(n + 255) / 256, 256, 0, stream>>>(images, imgN, n);
  }
  reorder_w<<<(27 * 64 + 255) / 256, 256, 0, stream>>>(cw1, w1T, 3, 64, 27 * 64);
  reorder_w<<<(576 * 128 + 255) / 256, 256, 0, stream>>>(cw2, w2T, 64, 128, 576 * 128);
  reorder_w<<<(1152 * 256 + 255) / 256, 256, 0, stream>>>(cw3, w3T, 128, 256, 1152 * 256);
  reorder_w<<<(2304 * 512 + 255) / 256, 256, 0, stream>>>(cw4, w4T, 256, 512, 2304 * 512);
  cvt_f32_bf16<<<(512 * 512 + 255) / 256, 256, 0, stream>>>(rnn_w, rnnB, 512 * 512);
  cvt_f32_bf16<<<(1024 * 512 + 255) / 256, 256, 0, stream>>>(fc1_w, fc1B, 1024 * 512);
  cvt_f32_bf16<<<(512 * 512 + 255) / 256, 256, 0, stream>>>(fc2_w, fc2B, 512 * 512);
  cvt_f32_bf16<<<(512 * 1000 + 255) / 256, 256, 0, stream>>>(clf_w, clfB, 512 * 1000);

  // ---- conv stack (implicit GEMM on WMMA), M tiles of 256 ----
  conv_wmma_kernel<3, 64, 224, 224><<<dim3(3136, 1), 256, 0, stream>>>(imgN, w1T, cb1, act1);
  conv_wmma_kernel<64, 128, 112, 112><<<dim3(784, 2), 256, 0, stream>>>(act1, w2T, cb2, act2);
  conv_wmma_kernel<128, 256, 56, 56><<<dim3(196, 4), 256, 0, stream>>>(act2, w3T, cb3, act3);
  conv_wmma_kernel<256, 512, 28, 28><<<dim3(49, 8), 256, 0, stream>>>(act3, w4T, cb4, act4);

  // ---- heads ----
  avgpool_kernel<<<64, 512, 0, stream>>>(act4, feats);  // feats[:, 0:512]
  embed_mean_kernel<<<64, 256, 0, stream>>>(emb, seqs, lens, meanB);
  gemm_wmma_kernel<false, true><<<dim3(1, 8), 256, 0, stream>>>(
      meanB, 512, rnnB, 512, rnn_b, feats + 512, 1024, 64, 512, 512);

  // ---- MLP ----
  gemm_wmma_kernel<true, true><<<dim3(1, 8), 256, 0, stream>>>(
      feats, 1024, fc1B, 512, fc1_b, h1B, 512, 64, 512, 1024);
  gemm_wmma_kernel<true, true><<<dim3(1, 8), 256, 0, stream>>>(
      h1B, 512, fc2B, 512, fc2_b, h2B, 512, 64, 512, 512);
  gemm_wmma_kernel<false, false><<<dim3(1, 16), 256, 0, stream>>>(
      h2B, 512, clfB, 1000, clf_b, (void*)out, 1000, 64, 1000, 512);
}
